// Codebook_47768626266382
// MI455X (gfx1250) — matrix operand order repaired and checked by reading, not compiled, not run on hardware
//
#include <hip/hip_runtime.h>
#include <stddef.h>

// ---------------------------------------------------------------------------
// CDNA5 (gfx1250) top-k similarity search:
//   cos  = normalize(z_rot) @ z_rot_book^T          -> top-20 / row
//   m2   = 2*z_trans@book^T - ||z||^2 - ||book||^2  -> top-20 / row
// GEMM via v_wmma_f32_16x16x32_bf16 with hi/lo split-bf16 (3 WMMAs/chunk,
// ~2^-16 relative error => f32-grade ranking). Books are pre-split once into
// bf16 hi/lo planes in workspace (amortizes conversion across all 128
// row-tile workgroups; planes are L2-resident: ~51MB << 192MB L2).
// 3 independent accumulators give 3-way ILP in the matrix pipe.
// ---------------------------------------------------------------------------

typedef __attribute__((ext_vector_type(16))) __bf16 v16bf;
typedef __attribute__((ext_vector_type(8)))  float  v8f;

constexpr int BATCH = 2048;
constexpr int NBOOK = 50000;
constexpr int DIM   = 128;
constexpr int TOPK  = 20;
constexpr float NEG_BIG = -3.402823466e38f;

// --- ||book_row||^2 for the euclidean branch ------------------------------
__global__ void rownorm2_kernel(const float* __restrict__ book,
                                float* __restrict__ bq) {
    int n = blockIdx.x * blockDim.x + threadIdx.x;
    if (n >= NBOOK) return;
    const float4* r = (const float4*)(book + (size_t)n * DIM);
    float acc = 0.f;
#pragma unroll 8
    for (int i = 0; i < DIM / 4; ++i) {
        float4 v = r[i];
        acc += v.x * v.x + v.y * v.y + v.z * v.z + v.w * v.w;
    }
    bq[n] = acc;
}

// --- one-shot f32 -> bf16 hi/lo split of a book ---------------------------
__global__ void split_bf16_kernel(const float* __restrict__ src,
                                  unsigned short* __restrict__ hi,
                                  unsigned short* __restrict__ lo,
                                  int nelem4) {
    int i = blockIdx.x * blockDim.x + threadIdx.x;
    if (i >= nelem4) return;
    float4 v = ((const float4*)src)[i];
    __bf16 h0 = (__bf16)v.x, h1 = (__bf16)v.y,
           h2 = (__bf16)v.z, h3 = (__bf16)v.w;
    __bf16 l0 = (__bf16)(v.x - (float)h0), l1 = (__bf16)(v.y - (float)h1),
           l2 = (__bf16)(v.z - (float)h2), l3 = (__bf16)(v.w - (float)h3);
    ushort4 hs, ls;
    hs.x = __builtin_bit_cast(unsigned short, h0);
    hs.y = __builtin_bit_cast(unsigned short, h1);
    hs.z = __builtin_bit_cast(unsigned short, h2);
    hs.w = __builtin_bit_cast(unsigned short, h3);
    ls.x = __builtin_bit_cast(unsigned short, l0);
    ls.y = __builtin_bit_cast(unsigned short, l1);
    ls.z = __builtin_bit_cast(unsigned short, l2);
    ls.w = __builtin_bit_cast(unsigned short, l3);
    ((ushort4*)hi)[i] = hs;
    ((ushort4*)lo)[i] = ls;
}

// --- main fused GEMM + top-k kernel ---------------------------------------
// MODE 0: cosine (A scaled by 1/||z||, score used directly, 1-wide labels)
// MODE 1: euclid (A scaled by 2, score - zq - bq, 3-wide labels)
// PRECONV: B fragments loaded from pre-split bf16 planes (fast path) vs.
//          converted from f32 in-loop (fallback when ws too small).
template <int MODE, bool PRECONV>
__global__ __launch_bounds__(32) void topk_wmma_kernel(
    const float* __restrict__ z,              // [BATCH, DIM]
    const float* __restrict__ book,           // [NBOOK, DIM] f32 (fallback)
    const unsigned short* __restrict__ bookhi,// [NBOOK, DIM] bf16 hi plane
    const unsigned short* __restrict__ booklo,// [NBOOK, DIM] bf16 lo plane
    const float* __restrict__ bq,             // [NBOOK] (MODE==1 only)
    const float* __restrict__ labsrc,         // [NBOOK, 1 or 3]
    float* __restrict__ vals_out,             // [BATCH, TOPK]
    int*   __restrict__ inds_out,             // [BATCH, TOPK]
    float* __restrict__ labs_out)             // [BATCH, TOPK, 1 or 3]
{
    __shared__ float s_scores[16 * 16];       // one 16x16 WMMA D tile
    __shared__ float s_corr[16];              // 1/||z|| (cos) or ||z||^2 (euc)
    __shared__ float s_vals[32 * TOPK];       // per-lane sorted ascending lists
    __shared__ int   s_idx [32 * TOPK];

    const int lane    = threadIdx.x;          // 0..31, single wave per block
    const int rowbase = blockIdx.x * 16;      // 16 batch rows per block

    // per-row norm^2 (lanes 0..15, one row each)
    if (lane < 16) {
        const float4* zr = (const float4*)(z + (size_t)(rowbase + lane) * DIM);
        float acc = 0.f;
#pragma unroll
        for (int i = 0; i < DIM / 4; ++i) {
            float4 v = zr[i];
            acc += v.x * v.x + v.y * v.y + v.z * v.z + v.w * v.w;
        }
        s_corr[lane] = (MODE == 0) ? rsqrtf(acc) : acc;
    }
#pragma unroll
    for (int k = 0; k < TOPK; ++k) {
        s_vals[lane * TOPK + k] = NEG_BIG;
        s_idx [lane * TOPK + k] = 0;
    }
    __syncthreads();

    // ---- A fragments, built once (contiguous k order, same as B below):
    // lane m=L&15, khalf=L>>4; element e -> K = 32*kc + 16*khalf + e
    const int   m      = lane & 15;
    const int   khalf  = lane >> 4;
    const float* zrow  = z + (size_t)(rowbase + m) * DIM;
    const float scaleA = (MODE == 0) ? s_corr[m] : 2.0f;
    v16bf ah[4], al[4];
#pragma unroll
    for (int kc = 0; kc < 4; ++kc) {
#pragma unroll
        for (int e = 0; e < 16; ++e) {
            int   basek = e;
            float v = zrow[32 * kc + basek + 16 * khalf] * scaleA;
            __bf16 h = (__bf16)v;
            ah[kc][e] = h;
            al[kc][e] = (__bf16)(v - (float)h);   // residual for split product
        }
    }

    const int   nlane   = lane & 15;      // B column owned by this lane
    const int   myrow   = lane >> 1;      // top-k: 2 lanes per row
    const int   colbase = (lane & 1) * 8;
    const float corr    = (MODE == 1) ? s_corr[myrow] : 0.f;
    const int   lb      = lane * TOPK;

    for (int nbase = 0; nbase < NBOOK; nbase += 16) {
        const size_t boff = (size_t)(nbase + nlane) * DIM;
        // 3 independent accumulator chains -> 3-way WMMA ILP
        v8f ca = {0.f, 0.f, 0.f, 0.f, 0.f, 0.f, 0.f, 0.f};
        v8f cb = {0.f, 0.f, 0.f, 0.f, 0.f, 0.f, 0.f, 0.f};
        v8f cc = {0.f, 0.f, 0.f, 0.f, 0.f, 0.f, 0.f, 0.f};
#pragma unroll
        for (int kc = 0; kc < 4; ++kc) {
            // B 32x16 layout: lane n=L&15, K = 32*kc + 16*(L>>4) + e
            // -> 16 consecutive elements per lane (32B bf16 / 64B f32)
            v16bf bh, bl;
            if (PRECONV) {
                bh = *(const v16bf*)(bookhi + boff + 32 * kc + 16 * khalf);
                bl = *(const v16bf*)(booklo + boff + 32 * kc + 16 * khalf);
            } else {
                const float* brow = book + boff + 32 * kc + 16 * khalf;
#pragma unroll
                for (int e = 0; e < 16; ++e) {
                    float v = brow[e];
                    __bf16 h = (__bf16)v;
                    bh[e] = h;
                    bl[e] = (__bf16)(v - (float)h);
                }
            }
            // split-bf16 product: hi*hi + hi*lo + lo*hi  (drop lo*lo)
            ca = __builtin_amdgcn_wmma_f32_16x16x32_bf16(
                     false, ah[kc], false, bh, (short)0, ca, false, false);
            cb = __builtin_amdgcn_wmma_f32_16x16x32_bf16(
                     false, ah[kc], false, bl, (short)0, cb, false, false);
            cc = __builtin_amdgcn_wmma_f32_16x16x32_bf16(
                     false, al[kc], false, bh, (short)0, cc, false, false);
        }

        // prefetch next book tile (global_prefetch_b8)
        if (nbase + 16 < NBOOK) {
            if (PRECONV) {
                __builtin_prefetch(
                    bookhi + boff + (size_t)16 * DIM + 16 * khalf, 0, 0);
                __builtin_prefetch(
                    booklo + boff + (size_t)16 * DIM + 16 * khalf, 0, 0);
            } else {
                __builtin_prefetch(
                    book + boff + (size_t)16 * DIM + 16 * khalf, 0, 0);
            }
        }

        // scatter C (D layout: VGPR r, lane L -> M = 8*(L>>4)+r, N = L&15)
#pragma unroll
        for (int r = 0; r < 8; ++r)
            s_scores[(khalf * 8 + r) * 16 + nlane] = ca[r] + cb[r] + cc[r];
        __syncthreads();

        // top-k maintenance: 2 lanes/row, 8 candidates each
        for (int j = 0; j < 8; ++j) {
            int   n = colbase + j;
            float s = s_scores[myrow * 16 + n];
            int   g = nbase + n;
            float v = (MODE == 0) ? s : (s - corr - bq[g]);
            if (v > s_vals[lb]) {               // beats current min -> insert
                int p = 0;
                while (p < TOPK - 1 && s_vals[lb + p + 1] < v) {
                    s_vals[lb + p] = s_vals[lb + p + 1];
                    s_idx [lb + p] = s_idx [lb + p + 1];
                    ++p;
                }
                s_vals[lb + p] = v;
                s_idx [lb + p] = g;
            }
        }
        __syncthreads();
    }

    // ---- final 40 -> 20 descending selection merge, one lane per row -----
    if (lane < 16) {
        const int base = (2 * lane) * TOPK;   // lanes 2r, 2r+1 are adjacent
        const int gb   = rowbase + lane;
        for (int k = 0; k < TOPK; ++k) {
            float bv = NEG_BIG;
            int   bi = 0x7fffffff, bp = 0;
            for (int j = 0; j < 2 * TOPK; ++j) {
                float v  = s_vals[base + j];
                int   gi = s_idx [base + j];
                if (v > bv || (v == bv && gi < bi)) { bv = v; bi = gi; bp = j; }
            }
            s_vals[base + bp] = NEG_BIG;      // consume
            vals_out[gb * TOPK + k] = bv;
            inds_out[gb * TOPK + k] = bi;
            if (MODE == 0) {
                labs_out[gb * TOPK + k] = labsrc[bi];
            } else {
                labs_out[(gb * TOPK + k) * 3 + 0] = labsrc[bi * 3 + 0];
                labs_out[(gb * TOPK + k) * 3 + 1] = labsrc[bi * 3 + 1];
                labs_out[(gb * TOPK + k) * 3 + 2] = labsrc[bi * 3 + 2];
            }
        }
    }
}

// ---------------------------------------------------------------------------
extern "C" void kernel_launch(void* const* d_in, const int* in_sizes, int n_in,
                              void* d_out, int out_size, void* d_ws,
                              size_t ws_size, hipStream_t stream) {
    const float* z_rot        = (const float*)d_in[0];  // [2048,128]
    const float* z_trans      = (const float*)d_in[1];  // [2048,128]
    const float* z_rot_book   = (const float*)d_in[2];  // [50000,128] (renormed)
    const float* z_trans_book = (const float*)d_in[3];  // [50000,128]
    const float* rot_book     = (const float*)d_in[4];  // [50000,1]
    const float* trans_book   = (const float*)d_in[5];  // [50000,3]

    float* out = (float*)d_out;
    // flat output layout (return order):
    const int BK = BATCH * TOPK;                 // 40960
    float* vals_cos = out;                       // [B,K]
    int*   ind_cos  = (int*)(out + BK);          // [B,K] int32 bit pattern
    float* lab_rot  = out + 2 * BK;              // [B,K,1]
    float* vals_euc = out + 3 * BK;              // [B,K]
    int*   ind_euc  = (int*)(out + 4 * BK);      // [B,K]
    float* lab_trn  = out + 5 * BK;              // [B,K,3]

    // workspace layout
    const size_t bqBytes   = ((size_t)NBOOK * sizeof(float) + 255) & ~(size_t)255;
    const size_t planeElts = (size_t)NBOOK * DIM;
    const size_t planeByte = planeElts * sizeof(unsigned short);
    const size_t needBytes = bqBytes + 4 * planeByte;   // ~51.4 MB

    float* bq = (float*)d_ws;                    // [50000] ||book_row||^2
    rownorm2_kernel<<<(NBOOK + 127) / 128, 128, 0, stream>>>(z_trans_book, bq);

    if (ws_size >= needBytes) {
        // fast path: pre-split both books into bf16 hi/lo planes once
        unsigned short* rot_hi = (unsigned short*)((char*)d_ws + bqBytes);
        unsigned short* rot_lo = rot_hi + planeElts;
        unsigned short* trn_hi = rot_lo + planeElts;
        unsigned short* trn_lo = trn_hi + planeElts;
        const int n4 = NBOOK * DIM / 4;
        split_bf16_kernel<<<(n4 + 255) / 256, 256, 0, stream>>>(
            z_rot_book, rot_hi, rot_lo, n4);
        split_bf16_kernel<<<(n4 + 255) / 256, 256, 0, stream>>>(
            z_trans_book, trn_hi, trn_lo, n4);

        topk_wmma_kernel<0, true><<<BATCH / 16, 32, 0, stream>>>(
            z_rot, z_rot_book, rot_hi, rot_lo, nullptr, rot_book,
            vals_cos, ind_cos, lab_rot);
        topk_wmma_kernel<1, true><<<BATCH / 16, 32, 0, stream>>>(
            z_trans, z_trans_book, trn_hi, trn_lo, bq, trans_book,
            vals_euc, ind_euc, lab_trn);
    } else {
        // fallback: convert B fragments from f32 in-loop
        topk_wmma_kernel<0, false><<<BATCH / 16, 32, 0, stream>>>(
            z_rot, z_rot_book, nullptr, nullptr, nullptr, rot_book,
            vals_cos, ind_cos, lab_rot);
        topk_wmma_kernel<1, false><<<BATCH / 16, 32, 0, stream>>>(
            z_trans, z_trans_book, nullptr, nullptr, bq, trans_book,
            vals_euc, ind_euc, lab_trn);
    }
}
